// WSDModel_78048145703062
// MI455X (gfx1250) — matrix operand order repaired and checked by reading, not compiled, not run on hardware
//
#include <hip/hip_runtime.h>
#include <cstdint>

// ---------------------------------------------------------------------------
// Problem constants (from reference setup_inputs)
// ---------------------------------------------------------------------------
constexpr int BB = 4;        // batch
constexpr int NN = 2048;     // sequence
constexpr int DD = 1024;     // model dim
constexpr int YY = 32000;    // output vocab
constexpr float NEG = -10000.0f;
constexpr float LN_EPS = 1e-5f;

typedef __attribute__((ext_vector_type(16))) _Float16 h16v;
typedef __attribute__((ext_vector_type(8)))  float    f8v;

// Async DMA: global -> LDS, 16B per lane, tracked by ASYNCcnt (ISA 08 §4).
__device__ __forceinline__ void async_ld_b128(uint32_t lds_addr, const void* gaddr) {
    asm volatile("global_load_async_to_lds_b128 %0, %1, off"
                 :: "v"(lds_addr), "v"(gaddr) : "memory");
}

// ---------------------------------------------------------------------------
// Elementwise helpers
// ---------------------------------------------------------------------------
__global__ __launch_bounds__(256)
void embed_kernel(const int* __restrict__ M_s, const float* __restrict__ E_v,
                  float* __restrict__ Xf, _Float16* __restrict__ Xh) {
    const int row = blockIdx.x;                 // b*N + n
    const int tok = M_s[row];
    const int d   = threadIdx.x * 4;            // 256 threads * 4 = 1024 = D
    const float4 v = *(const float4*)(E_v + (size_t)tok * DD + d);
    *(float4*)(Xf + (size_t)row * DD + d) = v;
    _Float16* xh = Xh + (size_t)row * DD + d;
    xh[0] = (_Float16)v.x; xh[1] = (_Float16)v.y;
    xh[2] = (_Float16)v.z; xh[3] = (_Float16)v.w;
}

__global__ __launch_bounds__(256)
void cvt_f32_f16(const float* __restrict__ src, _Float16* __restrict__ dst) {
    const size_t i = ((size_t)blockIdx.x * 256 + threadIdx.x) * 4;
    const float4 v = *(const float4*)(src + i);
    dst[i + 0] = (_Float16)v.x; dst[i + 1] = (_Float16)v.y;
    dst[i + 2] = (_Float16)v.z; dst[i + 3] = (_Float16)v.w;
}

__device__ __forceinline__ float block_reduce_sum(float x, float* red) {
    const int t = threadIdx.x;
    red[t] = x; __syncthreads();
    for (int s = 128; s > 0; s >>= 1) {
        if (t < s) red[t] += red[t + s];
        __syncthreads();
    }
    const float r = red[0]; __syncthreads();
    return r;
}

__device__ __forceinline__ float block_reduce_max(float x, float* red) {
    const int t = threadIdx.x;
    red[t] = x; __syncthreads();
    for (int s = 128; s > 0; s >>= 1) {
        if (t < s) red[t] = fmaxf(red[t], red[t + s]);
        __syncthreads();
    }
    const float r = red[0]; __syncthreads();
    return r;
}

// mask (replace) + causal (add) + |i-j| bias (add) + softmax over k
__global__ __launch_bounds__(256)
void softmax_kernel(const float* __restrict__ Sraw, const int* __restrict__ M_s,
                    float* __restrict__ Aout, _Float16* __restrict__ Ph) {
    __shared__ float red[256];
    const int b = blockIdx.x >> 11;           // / 2048
    const int q = blockIdx.x & (NN - 1);
    const size_t base = ((size_t)b * NN + q) * NN;

    float v[8];
    float mx = -3.4e38f;
#pragma unroll
    for (int j = 0; j < 8; ++j) {
        const int k = threadIdx.x + j * 256;
        float s = Sraw[base + k];
        if (M_s[b * NN + k] == 0) s = NEG;        // key-padding: SET
        if (q < k) s += NEG;                      // causal: ADD
        s += fabsf((float)(q - k));               // relative bias: ADD
        v[j] = s;
        mx = fmaxf(mx, s);
    }
    mx = block_reduce_max(mx, red);

    float sum = 0.f;
#pragma unroll
    for (int j = 0; j < 8; ++j) { v[j] = __expf(v[j] - mx); sum += v[j]; }
    sum = block_reduce_sum(sum, red);
    const float inv = 1.0f / sum;

#pragma unroll
    for (int j = 0; j < 8; ++j) {
        const int k = threadIdx.x + j * 256;
        const float p = v[j] * inv;
        Aout[base + k] = p;
        Ph[base + k]   = (_Float16)p;
    }
}

// H = LayerNorm(O2 + X) -> f16
__global__ __launch_bounds__(256)
void ln_kernel(const float* __restrict__ O2, const float* __restrict__ Xf,
               const float* __restrict__ gamma, const float* __restrict__ beta,
               _Float16* __restrict__ Hh) {
    __shared__ float red[256];
    const size_t base = (size_t)blockIdx.x * DD;
    float v[4];
    float s = 0.f;
#pragma unroll
    for (int j = 0; j < 4; ++j) {
        const int d = threadIdx.x + j * 256;
        v[j] = O2[base + d] + Xf[base + d];
        s += v[j];
    }
    const float mu = block_reduce_sum(s, red) * (1.0f / DD);
    float vs = 0.f;
#pragma unroll
    for (int j = 0; j < 4; ++j) { const float dlt = v[j] - mu; vs += dlt * dlt; }
    const float var = block_reduce_sum(vs, red) * (1.0f / DD);
    const float rstd = rsqrtf(var + LN_EPS);
#pragma unroll
    for (int j = 0; j < 4; ++j) {
        const int d = threadIdx.x + j * 256;
        Hh[base + d] = (_Float16)((v[j] - mu) * rstd * gamma[d] + beta[d]);
    }
}

// ---------------------------------------------------------------------------
// Generic WMMA GEMM: C[M,Nc] = A[M,K] * B   (B is [K,Nc] or, if TRANS_B,
// stored row-major [Nc,K]).  f16 inputs, f32 accumulate via
// v_wmma_f32_16x16x32_f16.  Block tile 128x128, K-step 32, 8 waves (2x4),
// each wave owns a 64x32 subtile = 4x2 WMMA fragments.
// Staging: double-buffered LDS; A (and B when not transposed) streamed with
// global_load_async_to_lds_b128 (ASYNCcnt pipeline); transposed B goes
// global->VGPR->ds_store scatter.
// Fragment layouts per CDNA5 ISA 7.12.2:
//   A 16x32 : lane L -> row M=L&15 ; elems 0..7 -> K=kb..kb+7,
//             elems 8..15 -> K=kb+16.. , kb = (L<16 ? 0 : 8)
//   B 32x16 : lane L -> N=L&15 ; elems as A: K=kb+j, kb+16+j
//   C 16x16 : lane L -> N=L&15 ; elem e -> M=e+(L<16?0:8)
// ---------------------------------------------------------------------------
template <bool TRANS_B, bool OUT_F16>
__global__ __launch_bounds__(256)
void gemm_wmma(const _Float16* __restrict__ A, const _Float16* __restrict__ Bm,
               float* __restrict__ Cf, _Float16* __restrict__ Ch,
               int M, int Nc, int K, int lda, int ldb, int ldc,
               long long bsA, long long bsB, long long bsC) {
    __shared__ _Float16 sA[2][128 * 32];   // [buf][row][k]
    __shared__ _Float16 sB[2][32 * 128];   // [buf][k][n]

    const int tid  = threadIdx.x;
    const int lane = tid & 31;
    const int wave = tid >> 5;
    const int wm   = wave >> 2;         // 0..1 -> 64 rows each
    const int wn   = wave & 3;          // 0..3 -> 32 cols each
    const int lhalf = lane >> 4;        // 0/1
    const int l15   = lane & 15;

    const int m0 = blockIdx.y * 128;
    const int n0 = blockIdx.x * 128;
    const _Float16* Ab = A  + (long long)blockIdx.z * bsA;
    const _Float16* Bb = Bm + (long long)blockIdx.z * bsB;
    const size_t cOff  = (size_t)((long long)blockIdx.z * bsC);

    const int ksteps = K >> 5;

    // per-thread staging coordinates (constant across K tiles)
    const int arow = tid >> 1, ahalf = tid & 1;          // A: 128 rows x 2 halves
    const int bk   = tid >> 3, bseg  = tid & 7;          // B(NN): 32 k-rows x 8 segs
    const int bn   = tid >> 1, bhalf = tid & 1;          // B(NT): 128 n-rows x 2 halves

    auto stage = [&](int kt, int buf) {
        // ---- A tile 128x32: 2 async b128 per thread ----
        {
            const _Float16* src = Ab + (size_t)(m0 + arow) * lda + kt * 32 + ahalf * 16;
            _Float16* dst = &sA[buf][arow * 32 + ahalf * 16];
            async_ld_b128((uint32_t)(uintptr_t)dst,       src);
            async_ld_b128((uint32_t)(uintptr_t)(dst + 8), src + 8);
            __builtin_prefetch((const void*)(src + 64), 0, 1);   // tile kt+2
        }
        // ---- B tile 32x128 ----
        if constexpr (TRANS_B) {
            const _Float16* src = Bb + (size_t)(n0 + bn) * ldb + kt * 32 + bhalf * 16;
            _Float16 tmp[16];
            *(uint4*)(tmp)     = ((const uint4*)src)[0];
            *(uint4*)(tmp + 8) = ((const uint4*)src)[1];
#pragma unroll
            for (int j = 0; j < 16; ++j) sB[buf][(bhalf * 16 + j) * 128 + bn] = tmp[j];
        } else {
            const _Float16* src = Bb + (size_t)(kt * 32 + bk) * ldb + n0 + bseg * 16;
            _Float16* dst = &sB[buf][bk * 128 + bseg * 16];
            async_ld_b128((uint32_t)(uintptr_t)dst,       src);
            async_ld_b128((uint32_t)(uintptr_t)(dst + 8), src + 8);
        }
    };

    f8v acc[4][2] = {};

    stage(0, 0);
    for (int kt = 0; kt < ksteps; ++kt) {
        const int buf = kt & 1;
        const bool more = (kt + 1 < ksteps);
        if (more) stage(kt + 1, buf ^ 1);

        // Async loads complete in order: allow next tile's (4 NN / 2 NT) to
        // remain in flight, require current tile's to be done.
        if (more) {
            if constexpr (TRANS_B) asm volatile("s_wait_asynccnt 0x2" ::: "memory");
            else                   asm volatile("s_wait_asynccnt 0x4" ::: "memory");
        } else {
            asm volatile("s_wait_asynccnt 0x0" ::: "memory");
        }
        __syncthreads();   // all waves' tile-kt data visible in LDS

        // ---- load fragments ----
        h16v afr[4];
#pragma unroll
        for (int i = 0; i < 4; ++i) {
            const _Float16* p = &sA[buf][(wm * 64 + i * 16 + l15) * 32];
            const int kb = lhalf * 8;
#pragma unroll
            for (int j = 0; j < 8; ++j) {
                afr[i][j]     = p[kb + j];
                afr[i][j + 8] = p[kb + 16 + j];
            }
        }
        h16v bfr[2];
#pragma unroll
        for (int jn = 0; jn < 2; ++jn) {
            const _Float16* p = &sB[buf][wn * 32 + jn * 16 + l15];
            const int kb = lhalf * 8;
#pragma unroll
            for (int j = 0; j < 8; ++j) {
                bfr[jn][j]     = p[(kb + j) * 128];
                bfr[jn][j + 8] = p[(kb + 16 + j) * 128];
            }
        }

        // ---- 8x WMMA ----
#pragma unroll
        for (int i = 0; i < 4; ++i)
#pragma unroll
            for (int jn = 0; jn < 2; ++jn)
                acc[i][jn] = __builtin_amdgcn_wmma_f32_16x16x32_f16(
                    false, afr[i], false, bfr[jn], (short)0, acc[i][jn],
                    false, false);
        __syncthreads();   // buf free for reuse two iterations later
    }

    // ---- epilogue ----
#pragma unroll
    for (int i = 0; i < 4; ++i)
#pragma unroll
        for (int jn = 0; jn < 2; ++jn) {
            const int gn = n0 + wn * 32 + jn * 16 + l15;
#pragma unroll
            for (int e = 0; e < 8; ++e) {
                const int gm = m0 + wm * 64 + i * 16 + lhalf * 8 + e;
                const size_t idx = cOff + (size_t)gm * ldc + gn;
                if constexpr (OUT_F16) Ch[idx] = (_Float16)acc[i][jn][e];
                else                   Cf[idx] = acc[i][jn][e];
            }
        }
}

// ---------------------------------------------------------------------------
// Host-side orchestration
// ---------------------------------------------------------------------------
extern "C" void kernel_launch(void* const* d_in, const int* in_sizes, int n_in,
                              void* d_out, int out_size, void* d_ws, size_t ws_size,
                              hipStream_t stream) {
    (void)in_sizes; (void)n_in; (void)out_size; (void)ws_size;

    const int*   M_s   = (const int*)d_in[0];
    const float* E_v   = (const float*)d_in[1];
    const float* E_y   = (const float*)d_in[2];
    const float* W_A   = (const float*)d_in[3];
    const float* W_O   = (const float*)d_in[4];
    const float* gamma = (const float*)d_in[5];
    const float* beta  = (const float*)d_in[6];

    float* yout = (float*)d_out;                             // [B,N,Y]
    float* Aout = yout + (size_t)BB * NN * YY;               // [B,N,N]

    // workspace carve-out (aliased where lifetimes allow)
    char* ws = (char*)d_ws;
    size_t off = 0;
    auto alloc = [&](size_t bytes) { size_t o = off; off += (bytes + 255) & ~(size_t)255; return o; };
    const size_t oXf   = alloc((size_t)BB * NN * DD * 4);    // fp32 X (residual)
    const size_t oXh   = alloc((size_t)BB * NN * DD * 2);    // f16 X; later reused as Hh
    const size_t oWAh  = alloc((size_t)DD * DD * 2);
    const size_t oWOh  = alloc((size_t)DD * DD * 2);
    const size_t oEyh  = alloc((size_t)YY * DD * 2);
    const size_t oQWh  = alloc((size_t)BB * NN * DD * 2);    // f16 X*W_A; later reused as O1h
    const size_t oS    = alloc((size_t)BB * NN * NN * 4);    // raw scores; later reused as O2
    const size_t oPh   = alloc((size_t)BB * NN * NN * 2);    // f16 softmax probs

    float*    Xf  = (float*)(ws + oXf);
    _Float16* Xh  = (_Float16*)(ws + oXh);
    _Float16* WAh = (_Float16*)(ws + oWAh);
    _Float16* WOh = (_Float16*)(ws + oWOh);
    _Float16* Eyh = (_Float16*)(ws + oEyh);
    _Float16* QWh = (_Float16*)(ws + oQWh);
    float*    Sr  = (float*)(ws + oS);
    _Float16* Ph  = (_Float16*)(ws + oPh);
    _Float16* O1h = QWh;            // alias: QWh dead after scores GEMM
    float*    O2  = Sr;             // alias: Sraw dead after softmax
    _Float16* Hh  = Xh;             // alias: Xh dead after P*X GEMM

    // 1) gather + f16 conversions
    embed_kernel<<<BB * NN, 256, 0, stream>>>(M_s, E_v, Xf, Xh);
    cvt_f32_f16<<<(DD * DD) / 1024, 256, 0, stream>>>(W_A, WAh);
    cvt_f32_f16<<<(DD * DD) / 1024, 256, 0, stream>>>(W_O, WOh);
    cvt_f32_f16<<<(YY * DD) / 1024, 256, 0, stream>>>(E_y, Eyh);

    // 2) QW = X * W_A          [8192x1024] = [8192x1024][1024x1024]
    gemm_wmma<false, true><<<dim3(DD / 128, (BB * NN) / 128, 1), 256, 0, stream>>>(
        Xh, WAh, nullptr, QWh, BB * NN, DD, DD, DD, DD, DD, 0, 0, 0);

    // 3) S[b] = QW[b] * X[b]^T  per-batch [2048x2048]
    gemm_wmma<true, false><<<dim3(NN / 128, NN / 128, BB), 256, 0, stream>>>(
        QWh, Xh, Sr, nullptr, NN, NN, DD, DD, DD, NN,
        (long long)NN * DD, (long long)NN * DD, (long long)NN * NN);

    // 4) mask/causal/bias + softmax -> A (fp32 out) and P (f16)
    softmax_kernel<<<BB * NN, 256, 0, stream>>>(Sr, M_s, Aout, Ph);

    // 5) O1[b] = P[b] * X[b]    per-batch [2048x1024]
    gemm_wmma<false, true><<<dim3(DD / 128, NN / 128, BB), 256, 0, stream>>>(
        Ph, Xh, nullptr, O1h, NN, DD, NN, NN, DD, DD,
        (long long)NN * NN, (long long)NN * DD, (long long)NN * DD);

    // 6) O2 = O1 * W_O          [8192x1024]
    gemm_wmma<false, false><<<dim3(DD / 128, (BB * NN) / 128, 1), 256, 0, stream>>>(
        O1h, WOh, O2, nullptr, BB * NN, DD, DD, DD, DD, DD, 0, 0, 0);

    // 7) H = LN(O2 + X) -> f16
    ln_kernel<<<BB * NN, 256, 0, stream>>>(O2, Xf, gamma, beta, Hh);

    // 8) logits = H * E_y^T     [8192x32000]  (dominant GEMM, fp32 out)
    gemm_wmma<true, false><<<dim3(YY / 128, (BB * NN) / 128, 1), 256, 0, stream>>>(
        Hh, Eyh, yout, nullptr, BB * NN, YY, DD, DD, DD, YY, 0, 0, 0);
}
